// FiBiNET_67963562492436
// MI455X (gfx1250) — matrix operand order repaired and checked by reading, not compiled, not run on hardware
//
#include <hip/hip_runtime.h>

typedef __bf16 bf16_t;
typedef __attribute__((ext_vector_type(16))) __bf16 v16bf;
typedef __attribute__((ext_vector_type(8)))  float  v8f;
typedef __attribute__((ext_vector_type(4)))  unsigned int uv4;
typedef __attribute__((ext_vector_type(4)))  int    iv4;
typedef __attribute__((ext_vector_type(4)))  float  fv4;

#define B_SZ   4096
#define F_SZ   39
#define E_SZ   16
#define NUMN   13
#define NCAT   26
#define VOCAB  100000
#define REDD   13
#define PAIRS  741
#define KDIM   (2*PAIRS*E_SZ)   // 23712

#if defined(__has_builtin)
#if __has_builtin(__builtin_amdgcn_global_load_async_to_lds_b128) && \
    __has_builtin(__builtin_amdgcn_s_wait_asynccnt)
#define USE_ASYNC_LDS 1
#endif
#endif

// ---------------------------------------------------------------------------
// 1) x[b,f,:] = concat(num outer num_emb, gathered cat embeddings)
// ---------------------------------------------------------------------------
__global__ void build_x_kernel(const int* __restrict__ cat, const float* __restrict__ num,
                               const float* __restrict__ etab, const float* __restrict__ nemb,
                               float* __restrict__ x) {
  int idx = blockIdx.x * blockDim.x + threadIdx.x;      // one thread per (b,f)
  if (idx >= B_SZ * F_SZ) return;
  int b = idx / F_SZ, f = idx % F_SZ;
  fv4 out[4];
  if (f < NUMN) {
    float s = num[b * NUMN + f];
    const fv4* src = (const fv4*)(nemb + f * E_SZ);
    for (int i = 0; i < 4; i++) out[i] = s * src[i];
  } else {
    int c = f - NUMN;
    int row = cat[b * NCAT + c];
    const fv4* src = (const fv4*)(etab + ((size_t)c * VOCAB + (size_t)row) * E_SZ);
    for (int i = 0; i < 4; i++) out[i] = src[i];
  }
  fv4* dst = (fv4*)(x + (size_t)idx * E_SZ);
  for (int i = 0; i < 4; i++) dst[i] = out[i];
}

// ---------------------------------------------------------------------------
// 2) SENet: Z = mean(x,-1); A = relu(relu(Z@W1)@W2). One thread per row.
// ---------------------------------------------------------------------------
__global__ void senet_kernel(const float* __restrict__ x, const float* __restrict__ W1,
                             const float* __restrict__ W2, float* __restrict__ Aout) {
  int b = blockIdx.x * blockDim.x + threadIdx.x;
  if (b >= B_SZ) return;
  const float* xr = x + (size_t)b * F_SZ * E_SZ;
  float Z[F_SZ];
  for (int f = 0; f < F_SZ; f++) {
    float s = 0.f;
    for (int e = 0; e < E_SZ; e++) s += xr[f * E_SZ + e];
    Z[f] = s * (1.0f / E_SZ);
  }
  float t[REDD];
  for (int r = 0; r < REDD; r++) {
    float s = 0.f;
    for (int f = 0; f < F_SZ; f++) s += Z[f] * W1[f * REDD + r];
    t[r] = fmaxf(s, 0.f);
  }
  for (int f = 0; f < F_SZ; f++) {
    float s = 0.f;
    for (int r = 0; r < REDD; r++) s += t[r] * W2[r * F_SZ + f];
    Aout[(size_t)b * F_SZ + f] = fmaxf(s, 0.f);
  }
}

// ---------------------------------------------------------------------------
// 3) Bilinear interactions. proj = x@W_bil per field; se_bi = A_i*A_j*bi.
//    Writes hid in bf16 row-major (WMMA A-operand friendly).
// ---------------------------------------------------------------------------
__global__ void bilinear_kernel(const float* __restrict__ x, const float* __restrict__ Ase,
                                const float* __restrict__ Wbil, bf16_t* __restrict__ hid) {
  __shared__ float xrow[F_SZ * E_SZ];
  __shared__ float proj[F_SZ * E_SZ];
  __shared__ float arow[F_SZ];
  __shared__ int   pairIJ[PAIRS];
  int b = blockIdx.x;
  int tid = threadIdx.x;
  const float* xr = x + (size_t)b * F_SZ * E_SZ;
  for (int o = tid; o < F_SZ * E_SZ; o += blockDim.x) xrow[o] = xr[o];
  for (int o = tid; o < F_SZ; o += blockDim.x) arow[o] = Ase[(size_t)b * F_SZ + o];
  for (int p = tid; p < PAIRS; p += blockDim.x) {
    // pairs before row i (triu, k=1): off(i) = i*(2F-i-1)/2
    int i = 0;
    while ((i + 1) * (2 * F_SZ - (i + 1) - 1) / 2 <= p) i++;
    int j = p - i * (2 * F_SZ - i - 1) / 2 + i + 1;
    pairIJ[p] = (i << 8) | j;
  }
  __syncthreads();
  for (int o = tid; o < F_SZ * E_SZ; o += blockDim.x) {
    int f = o >> 4, e = o & 15;
    float s = 0.f;
    for (int k = 0; k < E_SZ; k++) s += xrow[(f << 4) + k] * Wbil[((f * E_SZ) + k) * E_SZ + e];
    proj[o] = s;
  }
  __syncthreads();
  bf16_t* h = hid + (size_t)b * KDIM;
  for (int o = tid; o < PAIRS * E_SZ; o += blockDim.x) {
    int p = o >> 4, e = o & 15;
    int ij = pairIJ[p]; int i = ij >> 8, j = ij & 255;
    float bi = proj[(i << 4) + e] * xrow[(j << 4) + e];
    float se = arow[i] * arow[j] * bi;
    h[o]                 = (bf16_t)se;   // se_bi block first
    h[PAIRS * E_SZ + o]  = (bf16_t)bi;   // plain bi block second
  }
}

// ---------------------------------------------------------------------------
// 4) Pack fp32 weights (K x N) into bf16 WMMA B-fragment layout.
//    Tile = 32(K) x 16(N) = 512 bf16 = 1KB, stored as two 512B half-regions:
//      half 0: lane l's b[0..7]  at  l*8      (elements)
//      half 1: lane l's b[8..15] at  256+l*8
//    where b[i] = W[kb*32 + i + 16*(l>>4)][nb*16 + (l&15)]  (contiguous k order, same as loadA)
//    Halves are 16B per lane -> conflict-free ds_load_b128.
// ---------------------------------------------------------------------------
__global__ void pack_w_kernel(const float* __restrict__ W, bf16_t* __restrict__ Wp,
                              int K, int N) {
  int q = blockIdx.x * blockDim.x + threadIdx.x;
  if (q >= K * N) return;
  int e    = q & 511;          // element within tile
  int tile = q >> 9;
  int half = e >> 8;
  int l    = (e >> 3) & 31;
  int i    = (half << 3) + (e & 7);
  int ntiles = N >> 4;
  int nb = tile % ntiles;
  int kb = tile / ntiles;
  int k = (kb << 5) + i + ((l >> 4) << 4);
  int n = (nb << 4) + (l & 15);
  Wp[q] = (bf16_t)W[(size_t)k * N + n];
}

// ---------------------------------------------------------------------------
// 5) bf16 WMMA GEMM + bias + relu.  8 waves/block, 128x64 tile.
//    B tiles staged through double-buffered LDS via async-to-LDS; A fragments
//    double-buffered in registers via 2x-unrolled k-loop (no register moves).
//    Inside a step only two B-fragment registers are live (b0/b1 pipeline).
// ---------------------------------------------------------------------------
__global__ void gemm_bf16_wmma(const bf16_t* __restrict__ A, const bf16_t* __restrict__ Bp,
                               const float* __restrict__ bias, bf16_t* __restrict__ out,
                               int K, int N, int relu) {
  __shared__ __align__(16) bf16_t bsh[2][2048];   // 2 x 4KB (4 tiles of 1KB)
  int tid  = threadIdx.x;
  int lane = tid & 31;
  int wave = tid >> 5;
  int rowBase = blockIdx.x * 128 + wave * 16;
  int colBase = blockIdx.y * 64;
  int h = lane >> 4;
  int m = rowBase + (lane & 15);
  int ntiles = N >> 4;
  int NK = K >> 5;

  const bf16_t* arow = A + (size_t)m * K;

  v8f acc0, acc1, acc2, acc3;
  for (int v = 0; v < 8; v++) { acc0[v] = 0.f; acc1[v] = 0.f; acc2[v] = 0.f; acc3[v] = 0.f; }

  // contiguous 2048-element (4KB) chunk of packed B for k-block kb
  auto stageB = [&](int buf, int kb) {
    const bf16_t* src = Bp + (((size_t)kb * ntiles + (colBase >> 4)) << 9) + tid * 8;
#ifdef USE_ASYNC_LDS
    __builtin_amdgcn_global_load_async_to_lds_b128(
        (__attribute__((address_space(1))) iv4*)src,
        (__attribute__((address_space(3))) iv4*)(&bsh[buf][tid * 8]),
        0, 0);
#else
    *(uv4*)(&bsh[buf][tid * 8]) = *(const uv4*)src;
#endif
  };
  // A fragment (16x32 bf16): a[i] = A[m][kk + i + 16h] (matches packed B)
  auto loadA = [&](int kk, v16bf& f) {
    uv4* av = (uv4*)&f;
    av[0] = *(const uv4*)(arow + kk + 16 * h);
    av[1] = *(const uv4*)(arow + kk + 8 + 16 * h);
  };
  auto ldB = [&](const bf16_t* bb, int t, v16bf& f) {
    uv4* bv = (uv4*)&f;
    bv[0] = *(const uv4*)(bb + t * 512 + lane * 8);        // b[0..7]
    bv[1] = *(const uv4*)(bb + t * 512 + 256 + lane * 8);  // b[8..15]
  };
#define WMMA_BF16(a, b, c) __builtin_amdgcn_wmma_f32_16x16x32_bf16( \
      false, (a), false, (b), (short)0, (c), false, false)

  // 2-deep fragment pipeline: only b0/b1 live at once
  auto computeStep = [&](const bf16_t* bb, const v16bf& a) {
    v16bf b0, b1;
    ldB(bb, 0, b0);
    ldB(bb, 1, b1);
    acc0 = WMMA_BF16(a, b0, acc0);
    ldB(bb, 2, b0);
    acc1 = WMMA_BF16(a, b1, acc1);
    ldB(bb, 3, b1);
    acc2 = WMMA_BF16(a, b0, acc2);
    acc3 = WMMA_BF16(a, b1, acc3);
  };
#ifdef USE_ASYNC_LDS
#define GEMM_SYNC() do { __builtin_amdgcn_s_wait_asynccnt(0); __syncthreads(); } while (0)
#else
#define GEMM_SYNC() __syncthreads()
#endif

  v16bf a0, a1;
  loadA(0, a0);
  a1 = a0;
  stageB(0, 0);

  for (int kb = 0; kb < NK; kb += 2) {
    // even phase: consume buf0 / a0, stage buf1 / a1 for kb+1
    GEMM_SYNC();
    if (kb + 1 < NK) {
      stageB(1, kb + 1);
      loadA((kb + 1) << 5, a1);
    }
    __builtin_prefetch(arow + (kb << 5) + 512, 0, 2);
    computeStep(bsh[0], a0);
    if (kb + 1 >= NK) break;
    // odd phase: consume buf1 / a1, stage buf0 / a0 for kb+2
    GEMM_SYNC();
    if (kb + 2 < NK) {
      stageB(0, kb + 2);
      loadA((kb + 2) << 5, a0);
    }
    computeStep(bsh[1], a1);
  }

  // epilogue: c[v] -> row rowBase + v + 8h, col colBase + t*16 + (lane&15)
  float accs[4][8];
  for (int v = 0; v < 8; v++) {
    accs[0][v] = acc0[v]; accs[1][v] = acc1[v];
    accs[2][v] = acc2[v]; accs[3][v] = acc3[v];
  }
  for (int t = 0; t < 4; t++) {
    int col = colBase + t * 16 + (lane & 15);
    float bcol = bias[col];
    for (int v = 0; v < 8; v++) {
      int r = rowBase + v + 8 * h;
      float val = accs[t][v] + bcol;
      if (relu) val = fmaxf(val, 0.f);
      out[(size_t)r * N + col] = (bf16_t)val;
    }
  }
}

// ---------------------------------------------------------------------------
// 6) Final 256 -> 1 projection, one wave32 per row.
// ---------------------------------------------------------------------------
__global__ void final_kernel(const bf16_t* __restrict__ h3, const float* __restrict__ Wout,
                             const float* __restrict__ bout, float* __restrict__ out) {
  int gid = blockIdx.x * blockDim.x + threadIdx.x;
  int row = gid >> 5;
  int lane = gid & 31;
  if (row >= B_SZ) return;
  const bf16_t* hr = h3 + (size_t)row * 256;
  float s = 0.f;
  for (int i = 0; i < 8; i++) {
    int e = lane * 8 + i;
    s += (float)hr[e] * Wout[e];
  }
  for (int off = 16; off > 0; off >>= 1) s += __shfl_down(s, off, 32);
  if (lane == 0) out[row] = s + bout[0];
}

// ---------------------------------------------------------------------------
extern "C" void kernel_launch(void* const* d_in, const int* in_sizes, int n_in,
                              void* d_out, int out_size, void* d_ws, size_t ws_size,
                              hipStream_t stream) {
  const int*   cat  = (const int*)  d_in[0];
  const float* num  = (const float*)d_in[1];
  const float* etab = (const float*)d_in[2];
  const float* nemb = (const float*)d_in[3];
  const float* Wse1 = (const float*)d_in[4];
  const float* Wse2 = (const float*)d_in[5];
  const float* Wbil = (const float*)d_in[6];
  const float* Wd1  = (const float*)d_in[7];
  const float* bd1  = (const float*)d_in[8];
  const float* Wd2  = (const float*)d_in[9];
  const float* bd2  = (const float*)d_in[10];
  const float* Wd3  = (const float*)d_in[11];
  const float* bd3  = (const float*)d_in[12];
  const float* Wout = (const float*)d_in[13];
  const float* bout = (const float*)d_in[14];
  float* out = (float*)d_out;

  char* ws = (char*)d_ws;
  size_t off = 0;
  auto alloc = [&](size_t bytes) -> char* {
    char* p = ws + off;
    off += (bytes + 255) & ~(size_t)255;
    return p;
  };
  float*  x    = (float*) alloc((size_t)B_SZ * F_SZ * E_SZ * 4);
  float*  Ase  = (float*) alloc((size_t)B_SZ * F_SZ * 4);
  bf16_t* hid  = (bf16_t*)alloc((size_t)B_SZ * KDIM * 2);
  bf16_t* wd1p = (bf16_t*)alloc((size_t)KDIM * 1024 * 2);
  bf16_t* h1   = (bf16_t*)alloc((size_t)B_SZ * 1024 * 2);
  bf16_t* wd2p = (bf16_t*)alloc((size_t)1024 * 512 * 2);
  bf16_t* h2   = (bf16_t*)alloc((size_t)B_SZ * 512 * 2);
  bf16_t* wd3p = (bf16_t*)alloc((size_t)512 * 256 * 2);
  bf16_t* h3   = (bf16_t*)alloc((size_t)B_SZ * 256 * 2);

  build_x_kernel<<<(B_SZ * F_SZ + 255) / 256, 256, 0, stream>>>(cat, num, etab, nemb, x);
  senet_kernel<<<B_SZ / 256, 256, 0, stream>>>(x, Wse1, Wse2, Ase);
  bilinear_kernel<<<B_SZ, 256, 0, stream>>>(x, Ase, Wbil, hid);

  pack_w_kernel<<<(KDIM * 1024 + 255) / 256, 256, 0, stream>>>(Wd1, wd1p, KDIM, 1024);
  pack_w_kernel<<<(1024 * 512 + 255) / 256, 256, 0, stream>>>(Wd2, wd2p, 1024, 512);
  pack_w_kernel<<<(512 * 256 + 255) / 256, 256, 0, stream>>>(Wd3, wd3p, 512, 256);

  gemm_bf16_wmma<<<dim3(B_SZ / 128, 1024 / 64), 256, 0, stream>>>(hid, wd1p, bd1, h1, KDIM, 1024, 1);
  gemm_bf16_wmma<<<dim3(B_SZ / 128,  512 / 64), 256, 0, stream>>>(h1,  wd2p, bd2, h2, 1024,  512, 1);
  gemm_bf16_wmma<<<dim3(B_SZ / 128,  256 / 64), 256, 0, stream>>>(h2,  wd3p, bd3, h3,  512,  256, 1);

  final_kernel<<<(B_SZ * 32) / 256, 256, 0, stream>>>(h3, Wout, bout, out);
}
